// MultiScaleWindowGridTransformer_84061099917694
// MI455X (gfx1250) — compile-verified, matrix operand order checked by reading, not run on hardware
//
#include <hip/hip_runtime.h>
#include <hip/hip_bf16.h>
#include <math.h>

typedef __attribute__((ext_vector_type(16))) _Float16 v16h;
typedef __attribute__((ext_vector_type(8)))  _Float16 v8h;
typedef __attribute__((ext_vector_type(8)))  float    v8f;

#define TT 131072              // B*H*W tokens
#define CCH 256

// Async DMA: global -> LDS, 16B per lane, tracked by ASYNCcnt (CDNA5)
#define ASYNC_LD_B128(ldsOff, gptr) \
  asm volatile("global_load_async_to_lds_b128 %0, %1, off" \
               :: "v"(ldsOff), "v"(gptr) : "memory")
#define WAIT_ASYNC() asm volatile("s_wait_asynccnt 0x0" ::: "memory")

static __device__ __forceinline__ v16h ld_frag(const _Float16* p) {
  v8h lo = *(const v8h*)p;
  v8h hi = *(const v8h*)(p + 16);
  v16h r;
#pragma unroll
  for (int i = 0; i < 8; ++i) { r[i] = lo[i]; r[8 + i] = hi[i]; }
  return r;
}

// ---------------------------------------------------------------------------
// f32 -> f16 weight convert (weights are [N,K] row-major already)
// ---------------------------------------------------------------------------
__global__ __launch_bounds__(256) void cvt_f32_f16(
    const float* __restrict__ in, _Float16* __restrict__ out, int n) {
  int i = blockIdx.x * 256 + threadIdx.x;
  if (i < n) out[i] = (_Float16)in[i];
}

// ---------------------------------------------------------------------------
// Fused depthwise 3x3/5x5/7x7 convs + bias -> concatenated f16 [T, 768]
// ---------------------------------------------------------------------------
__global__ __launch_bounds__(256) void dwconv_cat(
    const float* __restrict__ x,
    const float* __restrict__ w1, const float* __restrict__ b1,
    const float* __restrict__ w2, const float* __restrict__ b2,
    const float* __restrict__ w3, const float* __restrict__ b3,
    _Float16* __restrict__ cat) {
  long idx = (long)blockIdx.x * 256 + threadIdx.x;   // NCHW order
  int ww = idx & 127;
  long r = idx >> 7;
  int hh = (int)(r & 127); r >>= 7;
  int c  = (int)(r & 255);
  int b  = (int)(r >> 8);
  const float* xp = x + (((long)b * 256 + c) << 14);
  const float* f1 = w1 + c * 9;
  const float* f2 = w2 + c * 25;
  const float* f3 = w3 + c * 49;
  float s1 = b1[c], s2 = b2[c], s3 = b3[c];
#pragma unroll
  for (int dy = -3; dy <= 3; ++dy) {
    int y = hh + dy;
    if (y < 0 || y >= 128) continue;
#pragma unroll
    for (int dx = -3; dx <= 3; ++dx) {
      int xw = ww + dx;
      if (xw < 0 || xw >= 128) continue;
      float v = xp[y * 128 + xw];
      if (dy >= -1 && dy <= 1 && dx >= -1 && dx <= 1)
        s1 += v * f1[(dy + 1) * 3 + (dx + 1)];
      if (dy >= -2 && dy <= 2 && dx >= -2 && dx <= 2)
        s2 += v * f2[(dy + 2) * 5 + (dx + 2)];
      s3 += v * f3[(dy + 3) * 7 + (dx + 3)];
    }
  }
  long t = ((long)b << 14) + hh * 128 + ww;
  _Float16* cp = cat + t * 768 + c;
  cp[0]   = (_Float16)s1;
  cp[256] = (_Float16)s2;
  cp[512] = (_Float16)s3;
}

// ---------------------------------------------------------------------------
// Generic WMMA GEMM:  out[M,N] = A[M,K](f16) * Bw[N,K]^T(f16) + bias, opt GELU
// Block tile 128x64, 8 waves (4x2), each wave 2x2 of 16x16, K-step 32.
// Double-buffered LDS fed by global_load_async_to_lds_b128 (ASYNCcnt).
// ---------------------------------------------------------------------------
__global__ __launch_bounds__(256) void gemm_wmma_f16(
    const _Float16* __restrict__ A, const _Float16* __restrict__ Bw,
    const float* __restrict__ bias, float* __restrict__ outF,
    _Float16* __restrict__ outH, int M, int N, int K, int act) {
  __shared__ _Float16 As[2][128][40];
  __shared__ _Float16 Bs[2][64][40];
  const int tid   = threadIdx.x;
  const int lane  = tid & 31;
  const int wid   = tid >> 5;
  const int waveM = wid >> 1;
  const int waveN = wid & 1;
  const long mBase = (long)blockIdx.y * 128;
  const long nBase = (long)blockIdx.x * 64;
  const int lm    = lane & 15;
  const int khalf = (lane < 16) ? 0 : 8;

  // Per-thread 16B staging chunks: A rows rA0 / rA0+64, B row rB
  const int rA0 = tid >> 2, kA0 = (tid & 3) << 3;
  const int rB  = tid >> 2, kB  = (tid & 3) << 3;
  const _Float16* gA0 = A  + (mBase + rA0) * (long)K + kA0;
  const _Float16* gA1 = gA0 + 64 * (long)K;
  const _Float16* gB  = Bw + (nBase + rB) * (long)K + kB;
  const unsigned lA0[2] = { (unsigned)(uintptr_t)&As[0][rA0][kA0],
                            (unsigned)(uintptr_t)&As[1][rA0][kA0] };
  const unsigned lA1[2] = { (unsigned)(uintptr_t)&As[0][rA0 + 64][kA0],
                            (unsigned)(uintptr_t)&As[1][rA0 + 64][kA0] };
  const unsigned lB[2]  = { (unsigned)(uintptr_t)&Bs[0][rB][kB],
                            (unsigned)(uintptr_t)&Bs[1][rB][kB] };

  v8f zero = {};
  v8f acc[2][2];
  acc[0][0] = zero; acc[0][1] = zero; acc[1][0] = zero; acc[1][1] = zero;

  // prologue: stream first K-tile into buffer 0
  ASYNC_LD_B128(lA0[0], gA0);
  ASYNC_LD_B128(lA1[0], gA1);
  ASYNC_LD_B128(lB[0],  gB);

  const int nsteps = K >> 5;
  for (int s = 0; s < nsteps; ++s) {
    const int cur = s & 1;
    WAIT_ASYNC();          // this wave's DMA into buf[cur] done
    __syncthreads();       // every wave's DMA done; prev reads of buf[cur] done
    if (s + 1 < nsteps) {  // stream next tile into the other buffer
      const long koff = (long)(s + 1) << 5;
      ASYNC_LD_B128(lA0[1 - cur], gA0 + koff);
      ASYNC_LD_B128(lA1[1 - cur], gA1 + koff);
      ASYNC_LD_B128(lB[1 - cur],  gB  + koff);
      if (s + 2 < nsteps) {
        __builtin_prefetch(gB  + ((long)(s + 2) << 5), 0, 1);
        __builtin_prefetch(gA0 + ((long)(s + 2) << 5), 0, 1);
      }
    }
    v16h af[2], bf[2];
#pragma unroll
    for (int mi = 0; mi < 2; ++mi)
      af[mi] = ld_frag(&As[cur][waveM * 32 + mi * 16 + lm][khalf]);
#pragma unroll
    for (int ni = 0; ni < 2; ++ni)
      bf[ni] = ld_frag(&Bs[cur][waveN * 32 + ni * 16 + lm][khalf]);
#pragma unroll
    for (int mi = 0; mi < 2; ++mi)
#pragma unroll
      for (int ni = 0; ni < 2; ++ni)
        acc[mi][ni] = __builtin_amdgcn_wmma_f32_16x16x32_f16(
            false, af[mi], false, bf[ni], (short)0, acc[mi][ni], false, false);
    __syncthreads();       // all reads of buf[cur] done before it is re-filled
  }

  const int rowOff = (lane < 16) ? 0 : 8;
#pragma unroll
  for (int mi = 0; mi < 2; ++mi)
#pragma unroll
    for (int ni = 0; ni < 2; ++ni) {
      long col = nBase + waveN * 32 + ni * 16 + lm;
      float bv = bias ? bias[col] : 0.0f;
#pragma unroll
      for (int r = 0; r < 8; ++r) {
        long row = mBase + waveM * 32 + mi * 16 + r + rowOff;
        float v = acc[mi][ni][r] + bv;
        if (act) v = 0.5f * v * (1.0f + erff(v * 0.7071067811865475f));
        if (outF) outF[row * N + col] = v;
        else      outH[row * N + col] = (_Float16)v;
      }
    }
}

// ---------------------------------------------------------------------------
// LayerNorm over C=256 (wave per token) with optional window/grid permutation
// mode: 0=identity, 1=window order, 2=grid order. out is f16 [T,256].
// ---------------------------------------------------------------------------
__global__ __launch_bounds__(256) void ln_permute(
    const float* __restrict__ in, const float* __restrict__ g,
    const float* __restrict__ be, _Float16* __restrict__ out, int mode) {
  const int token = blockIdx.x * 8 + (threadIdx.x >> 5);
  const int lane  = threadIdx.x & 31;
  const float* row = in + (long)token * 256;
  float v[8];
  float s = 0.0f;
#pragma unroll
  for (int i = 0; i < 8; ++i) { v[i] = row[lane + 32 * i]; s += v[i]; }
#pragma unroll
  for (int d = 1; d < 32; d <<= 1) s += __shfl_xor(s, d, 32);
  float mean = s * (1.0f / 256.0f);
  float q = 0.0f;
#pragma unroll
  for (int i = 0; i < 8; ++i) { float dd = v[i] - mean; q += dd * dd; }
#pragma unroll
  for (int d = 1; d < 32; d <<= 1) q += __shfl_xor(q, d, 32);
  float rstd = rsqrtf(q * (1.0f / 256.0f) + 1e-5f);
  int b = token >> 14, hw = token & 16383;
  int hh = hw >> 7, ww = hw & 127;
  long dt;
  if (mode == 0) dt = token;
  else if (mode == 1)
    dt = ((((long)b * 16 + (hh >> 3)) * 16 + (ww >> 3)) << 6) + ((hh & 7) * 8 + (ww & 7));
  else
    dt = ((((long)b * 16 + (hh & 15)) * 16 + (ww & 15)) << 6) + ((hh >> 4) * 8 + (ww >> 4));
  _Float16* orow = out + dt * 256;
#pragma unroll
  for (int i = 0; i < 8; ++i) {
    int c = lane + 32 * i;
    orow[c] = (_Float16)((v[i] - mean) * rstd * g[c] + be[c]);
  }
}

// ---------------------------------------------------------------------------
// Window attention: one wave per (window, head). qkv f16 [T,768] in window
// order (q|k|v, head = 32 contiguous channels). out f16 [T,256].
// S = QK^T (16 wmma), register softmax via half-wave shfl, O = PV (16 wmma).
// ---------------------------------------------------------------------------
__global__ __launch_bounds__(64) void win_attn(
    const _Float16* __restrict__ qkv, _Float16* __restrict__ out) {
  __shared__ _Float16 Ps[2][64][72];
  const int win   = blockIdx.x;
  const int wslot = threadIdx.x >> 5;
  const int head  = blockIdx.y * 2 + wslot;
  const int lane  = threadIdx.x & 31;
  const int lm    = lane & 15;
  const int khalf = (lane < 16) ? 0 : 8;
  const int rowOff = (lane < 16) ? 0 : 8;
  const long rowBase = (long)win * 64;

  v16h qf[4], kf[4];
#pragma unroll
  for (int mi = 0; mi < 4; ++mi)
    qf[mi] = ld_frag(&qkv[(rowBase + mi * 16 + lm) * 768 + head * 32 + khalf]);
#pragma unroll
  for (int ni = 0; ni < 4; ++ni)
    kf[ni] = ld_frag(&qkv[(rowBase + ni * 16 + lm) * 768 + 256 + head * 32 + khalf]);

  v8f zero = {};
  v8f S[4][4];
#pragma unroll
  for (int mi = 0; mi < 4; ++mi)
#pragma unroll
    for (int ni = 0; ni < 4; ++ni)
      S[mi][ni] = __builtin_amdgcn_wmma_f32_16x16x32_f16(
          false, qf[mi], false, kf[ni], (short)0, zero, false, false);

  const float scale = 0.17677669529663687f;   // 1/sqrt(32)
#pragma unroll
  for (int mi = 0; mi < 4; ++mi) {
#pragma unroll
    for (int r = 0; r < 8; ++r) {
      float mx = -1e30f;
#pragma unroll
      for (int ni = 0; ni < 4; ++ni) mx = fmaxf(mx, S[mi][ni][r]);
#pragma unroll
      for (int d = 1; d < 16; d <<= 1) mx = fmaxf(mx, __shfl_xor(mx, d, 32));
      float sum = 0.0f;
#pragma unroll
      for (int ni = 0; ni < 4; ++ni) {
        float e = __expf((S[mi][ni][r] - mx) * scale);
        S[mi][ni][r] = e;
        sum += e;
      }
#pragma unroll
      for (int d = 1; d < 16; d <<= 1) sum += __shfl_xor(sum, d, 32);
      float inv = 1.0f / sum;
      int row = mi * 16 + r + rowOff;
#pragma unroll
      for (int ni = 0; ni < 4; ++ni)
        Ps[wslot][row][ni * 16 + lm] = (_Float16)(S[mi][ni][r] * inv);
    }
  }
  __syncthreads();

  v16h vf[2][2];
#pragma unroll
  for (int kk = 0; kk < 2; ++kk)
#pragma unroll
    for (int ni = 0; ni < 2; ++ni) {
      const int ch = 512 + head * 32 + ni * 16 + lm;
      const int kb = kk * 32 + khalf;
      v16h t;
#pragma unroll
      for (int i = 0; i < 8; ++i) {
        t[i]     = qkv[(rowBase + kb + i) * 768 + ch];
        t[8 + i] = qkv[(rowBase + kb + 16 + i) * 768 + ch];
      }
      vf[kk][ni] = t;
    }

  v8f O[4][2];
#pragma unroll
  for (int mi = 0; mi < 4; ++mi)
#pragma unroll
    for (int ni = 0; ni < 2; ++ni) O[mi][ni] = zero;
#pragma unroll
  for (int kk = 0; kk < 2; ++kk)
#pragma unroll
    for (int mi = 0; mi < 4; ++mi) {
      v16h a = ld_frag(&Ps[wslot][mi * 16 + lm][kk * 32 + khalf]);
#pragma unroll
      for (int ni = 0; ni < 2; ++ni)
        O[mi][ni] = __builtin_amdgcn_wmma_f32_16x16x32_f16(
            false, a, false, vf[kk][ni], (short)0, O[mi][ni], false, false);
    }

#pragma unroll
  for (int mi = 0; mi < 4; ++mi)
#pragma unroll
    for (int ni = 0; ni < 2; ++ni)
#pragma unroll
      for (int r = 0; r < 8; ++r) {
        int row = mi * 16 + r + rowOff;
        out[(rowBase + row) * 256 + head * 32 + ni * 16 + lm] = (_Float16)O[mi][ni][r];
      }
}

// ---------------------------------------------------------------------------
// Residual with window/grid reverse: out[t,c] = base + proj[perm(t), c]
// base: NCHW x (baseTok=0) or token-major (baseTok=1)
// ---------------------------------------------------------------------------
__global__ __launch_bounds__(256) void residual_reverse(
    const float* __restrict__ base, const float* __restrict__ proj,
    float* __restrict__ out, int mode, int baseTok) {
  long idx = (long)blockIdx.x * 256 + threadIdx.x;
  long t = idx >> 8;
  int c = (int)(idx & 255);
  int b = (int)(t >> 14);
  int hw = (int)(t & 16383);
  int hh = hw >> 7, ww = hw & 127;
  long st = (mode == 1)
      ? ((((long)b * 16 + (hh >> 3)) * 16 + (ww >> 3)) << 6) + ((hh & 7) * 8 + (ww & 7))
      : ((((long)b * 16 + (hh & 15)) * 16 + (ww & 15)) << 6) + ((hh >> 4) * 8 + (ww >> 4));
  float bv = baseTok ? base[idx] : base[(((long)b * 256 + c) << 14) + hw];
  out[idx] = bv + proj[st * 256 + c];
}

// final: out NCHW = x3(token-major) + mlp(token-major)
__global__ __launch_bounds__(256) void final_add_nchw(
    const float* __restrict__ xtok, const float* __restrict__ mlp,
    float* __restrict__ out) {
  long idx = (long)blockIdx.x * 256 + threadIdx.x;   // NCHW
  int hw = (int)(idx & 16383);
  long r = idx >> 14;
  int c = (int)(r & 255);
  int b = (int)(r >> 8);
  long t = (((long)b << 14) + hw) * 256 + c;
  out[idx] = xtok[t] + mlp[t];
}

// ---------------------------------------------------------------------------
extern "C" void kernel_launch(void* const* d_in, const int* in_sizes, int n_in,
                              void* d_out, int out_size, void* d_ws, size_t ws_size,
                              hipStream_t stream) {
  (void)in_sizes; (void)n_in; (void)out_size; (void)ws_size;
  const float* x      = (const float*)d_in[0];
  const float* w1 = (const float*)d_in[1];  const float* b1 = (const float*)d_in[2];
  const float* w2 = (const float*)d_in[3];  const float* b2 = (const float*)d_in[4];
  const float* w3 = (const float*)d_in[5];  const float* b3 = (const float*)d_in[6];
  const float* wf = (const float*)d_in[7];  const float* bf = (const float*)d_in[8];
  const float* gw = (const float*)d_in[9];  const float* bw = (const float*)d_in[10];
  const float* wqkv_w = (const float*)d_in[11]; const float* bqkv_w = (const float*)d_in[12];
  const float* wo_w   = (const float*)d_in[13]; const float* bo_w   = (const float*)d_in[14];
  const float* gg = (const float*)d_in[15]; const float* bg = (const float*)d_in[16];
  const float* wqkv_g = (const float*)d_in[17]; const float* bqkv_g = (const float*)d_in[18];
  const float* wo_g   = (const float*)d_in[19]; const float* bo_g   = (const float*)d_in[20];
  const float* gm = (const float*)d_in[21]; const float* bm = (const float*)d_in[22];
  const float* m1w = (const float*)d_in[23]; const float* m1b = (const float*)d_in[24];
  const float* m2w = (const float*)d_in[25]; const float* m2b = (const float*)d_in[26];

  char* ws = (char*)d_ws;
  _Float16* wide = (_Float16*)ws;                 // up to T*1024 f16 = 268 MB
  float*    f32a = (float*)(ws + 268435456);      // T*256 f32 = 134 MB
  _Float16* f16a = (_Float16*)(ws + 402653184);   // T*256 f16 = 67 MB
  float*    x2   = (float*)(ws + 469762048);      // 134 MB
  float*    x3   = (float*)(ws + 603979776);      // 134 MB
  _Float16* wts  = (_Float16*)(ws + 738197504);
  _Float16* wfT   = wts;
  _Float16* wqkvW = wfT + 196608;
  _Float16* woW   = wqkvW + 196608;
  _Float16* wqkvG = woW + 65536;
  _Float16* woG   = wqkvG + 196608;
  _Float16* m1h   = woG + 65536;
  _Float16* m2h   = m1h + 262144;

  dim3 blk(256);
  cvt_f32_f16<<<768, blk, 0, stream>>>(wf, wfT, 196608);
  cvt_f32_f16<<<768, blk, 0, stream>>>(wqkv_w, wqkvW, 196608);
  cvt_f32_f16<<<256, blk, 0, stream>>>(wo_w, woW, 65536);
  cvt_f32_f16<<<768, blk, 0, stream>>>(wqkv_g, wqkvG, 196608);
  cvt_f32_f16<<<256, blk, 0, stream>>>(wo_g, woG, 65536);
  cvt_f32_f16<<<1024, blk, 0, stream>>>(m1w, m1h, 262144);
  cvt_f32_f16<<<1024, blk, 0, stream>>>(m2w, m2h, 262144);

  // multi-scale depthwise convs -> cat f16 [T,768]
  dwconv_cat<<<TT * CCH / 256, blk, 0, stream>>>(x, w1, b1, w2, b2, w3, b3, wide);
  // fusion GEMM -> x_multi f32 [T,256]
  gemm_wmma_f16<<<dim3(4, TT / 128), blk, 0, stream>>>(wide, wfT, bf, f32a, nullptr, TT, 256, 768, 0);

  // ---- window branch ----
  ln_permute<<<TT / 8, blk, 0, stream>>>(f32a, gw, bw, f16a, 1);
  gemm_wmma_f16<<<dim3(12, TT / 128), blk, 0, stream>>>(f16a, wqkvW, bqkv_w, nullptr, wide, TT, 768, 256, 0);
  win_attn<<<dim3(2048, 4), dim3(64), 0, stream>>>(wide, f16a);
  gemm_wmma_f16<<<dim3(4, TT / 128), blk, 0, stream>>>(f16a, woW, bo_w, f32a, nullptr, TT, 256, 256, 0);
  residual_reverse<<<TT * CCH / 256, blk, 0, stream>>>(x, f32a, x2, 1, 0);

  // ---- grid branch ----
  ln_permute<<<TT / 8, blk, 0, stream>>>(x2, gg, bg, f16a, 2);
  gemm_wmma_f16<<<dim3(12, TT / 128), blk, 0, stream>>>(f16a, wqkvG, bqkv_g, nullptr, wide, TT, 768, 256, 0);
  win_attn<<<dim3(2048, 4), dim3(64), 0, stream>>>(wide, f16a);
  gemm_wmma_f16<<<dim3(4, TT / 128), blk, 0, stream>>>(f16a, woG, bo_g, f32a, nullptr, TT, 256, 256, 0);
  residual_reverse<<<TT * CCH / 256, blk, 0, stream>>>(x2, f32a, x3, 2, 1);

  // ---- MLP ----
  ln_permute<<<TT / 8, blk, 0, stream>>>(x3, gm, bm, f16a, 0);
  gemm_wmma_f16<<<dim3(16, TT / 128), blk, 0, stream>>>(f16a, m1h, m1b, nullptr, wide, TT, 1024, 256, 1);
  gemm_wmma_f16<<<dim3(4, TT / 128), blk, 0, stream>>>(wide, m2h, m2b, f32a, nullptr, TT, 256, 1024, 0);
  final_add_nchw<<<TT * CCH / 256, blk, 0, stream>>>(x3, f32a, (float*)d_out);
}
